// MultiHeadAttention_19533511262244
// MI455X (gfx1250) — matrix operand order repaired and checked by reading, not compiled, not run on hardware
//
#include <hip/hip_runtime.h>
#include <stdint.h>

#define D_MODEL 1024
#define NHEAD   16
#define HD      64
#define SEQ     2048
#define BATCH   2
#define MROWS   (BATCH*SEQ)     // 4096
#define NQKV    (3*D_MODEL)     // 3072
#define SCALE   0.125f          // 1/sqrt(64)

typedef __attribute__((ext_vector_type(16))) __bf16 v16bf;
typedef __attribute__((ext_vector_type(8)))  float  v8f;

union Frag16 { v16bf v; uint32_t u[8]; };

__device__ inline uint16_t f2bf(float f) {
  uint32_t u = __float_as_uint(f);
  u += 0x7FFFu + ((u >> 16) & 1u);   // round-to-nearest-even
  return (uint16_t)(u >> 16);
}

// A-matrix (16-bit, 16x32) K-pair base for VGPR r, lane-half h
__device__ inline int a_kpair(int r, int half) {
  return (r < 4 ? 2*r : 16 + 2*(r-4)) + 8*half;
}

// ---------------------------------------------------------------- conversions
__global__ void cvt_bf16(const float* __restrict__ src, uint16_t* __restrict__ dst, int n) {
  int i = blockIdx.x*blockDim.x + threadIdx.x;
  int stride = gridDim.x*blockDim.x;
  for (; i < n; i += stride) dst[i] = f2bf(src[i]);
}

// ---------------------------------------------------------------- GEMM core
// One wave computes a 16(M) x 64(N) tile: acc[j] is the j-th 16x16 sub-tile.
// A: [M x lda] bf16 row-major.  Bw: [K x ldb] bf16 row-major (K rows, N cols).
__device__ inline void gemm16x64_bf16(const uint16_t* __restrict__ A,
                                      const uint16_t* __restrict__ Bw,
                                      int lda, int ldb, int m0, int n0, int K,
                                      v8f* acc) {
  int lane = threadIdx.x & 31;
  int n15 = lane & 15, half = lane >> 4;
  const uint16_t* Arow = A + (size_t)(m0 + n15)*lda;
  for (int k0 = 0; k0 < K; k0 += 32) {
    Frag16 a;
#pragma unroll
    for (int r = 0; r < 8; ++r)
      a.u[r] = *(const uint32_t*)(Arow + k0 + a_kpair(r, half));
    // B fragment: lane holds N column (n15), VGPR r holds K pair a_kpair(r, half)
    const uint16_t* Brow = Bw + (size_t)k0*ldb + n0 + n15;
#pragma unroll
    for (int j = 0; j < 4; ++j) {
      Frag16 bb;
#pragma unroll
      for (int r = 0; r < 8; ++r) {
        int k = a_kpair(r, half);
        bb.u[r] = (uint32_t)Brow[(size_t)k*ldb + j*16] | ((uint32_t)Brow[(size_t)(k+1)*ldb + j*16] << 16);
      }
      acc[j] = __builtin_amdgcn_wmma_f32_16x16x32_bf16(
                 false, a.v, false, bb.v, (short)0, acc[j], false, false);
    }
  }
}

// ---------------------------------------------------------------- QKV projection
// qkv = x @ Wqkv + bqkv, scattered as:
//   Q  [bh, t, d]  natural    (A fragments of the score GEMM)
//   Kt [bh, d, t]  transposed (B fragments of the score GEMM: contiguous key pairs)
//   V  [bh, t, d]  natural    (B fragments of P@V: contiguous d pairs)
__global__ void __launch_bounds__(128)
qkv_gemm(const uint16_t* __restrict__ xbf, const uint16_t* __restrict__ wbf,
         const float* __restrict__ bias,
         uint16_t* __restrict__ Qb, uint16_t* __restrict__ Kt, uint16_t* __restrict__ Vb) {
  int wave = threadIdx.x >> 5;
  int wtile = blockIdx.x*4 + wave;       // 256 m-tiles x 48 n-strips = 12288
  int tm = wtile & 255;
  int tn = wtile >> 8;
  int m0 = tm*16, n0 = tn*64;

  v8f acc[4] = {};
  gemm16x64_bf16(xbf, wbf, D_MODEL, NQKV, m0, n0, D_MODEL, acc);

  int lane = threadIdx.x & 31;
  int n15 = lane & 15, half = lane >> 4;
  int sel = n0 / D_MODEL;                // 0=Q 1=K 2=V (strip is 64-aligned)
  int h   = (n0 % D_MODEL) / HD;
#pragma unroll
  for (int j = 0; j < 4; ++j) {
    int d = j*16 + n15;
    float bj = bias[n0 + j*16 + n15];
#pragma unroll
    for (int r = 0; r < 8; ++r) {
      int m = m0 + r + 8*half;
      int b = m / SEQ, t = m % SEQ;
      int bh = b*NHEAD + h;
      uint16_t val = f2bf(acc[j][r] + bj);
      if (sel == 0)      Qb[((size_t)bh*SEQ + t)*HD + d] = val;
      else if (sel == 1) Kt[((size_t)bh*HD + d)*SEQ + t] = val;
      else               Vb[((size_t)bh*SEQ + t)*HD + d] = val;
    }
  }
}

// ---------------------------------------------------------------- fused attention
// One block (4 waves, 128 thr) per (bh, 16-row query tile).
// LDS: sc[16][2048] f32 scores (128KB). After softmax, bf16 probabilities are
// packed IN PLACE into the same buffer: bf16 index (row, s) lives at
//   uint16 offset  row*4096 + (s>>8)*512 + (s&255)
// i.e. the first half of each 256-col fp32 segment. 128KB -> 2 blocks/WGP.
__global__ void __launch_bounds__(128)
attn_kernel(const uint16_t* __restrict__ Qb, const uint16_t* __restrict__ Kt,
            const uint16_t* __restrict__ Vb,
            float* __restrict__ wout, uint16_t* __restrict__ attnbf) {
  extern __shared__ char smem[];
  float*    sc  = (float*)smem;
  uint16_t* pb  = (uint16_t*)smem;       // aliased, packed layout (see above)
  __shared__ float red[128];

  int bh   = blockIdx.x >> 7;            // 32 (b,h) pairs
  int tile = blockIdx.x & 127;           // 128 row tiles
  int t0   = tile * 16;
  int b = bh / NHEAD, h = bh % NHEAD;
  int tid = threadIdx.x;
  int wave = tid >> 5, lane = tid & 31;
  int n15 = lane & 15, half = lane >> 4;
  int chan0 = n15 + 16*half;             // B-fragment K index

  // Q A-fragments (hd=64 -> two K=32 fragments); each wave loads its own copy.
  Frag16 q0, q1;
  {
    const uint16_t* Qrow = Qb + ((size_t)bh*SEQ + t0 + n15)*HD;
#pragma unroll
    for (int r = 0; r < 8; ++r) {
      int k = a_kpair(r, half);
      q0.u[r] = *(const uint32_t*)(Qrow + k);
      q1.u[r] = *(const uint32_t*)(Qrow + 32 + k);
    }
  }

  // ---- Phase 1: S = Q K^T * scale -> LDS (waves split the 128 column tiles)
  const uint16_t* K0 = Kt + ((size_t)bh*HD)*SEQ + n15;
  const uint16_t* K1 = Kt + ((size_t)bh*HD + 32)*SEQ + n15;
  for (int ct = wave; ct < SEQ/16; ct += 4) {
    int s0 = ct*16;
    __builtin_prefetch(K0 + s0 + 64, 0, 0);   // next tile for this wave
    __builtin_prefetch(K1 + s0 + 64, 0, 0);
    Frag16 b0, b1;
#pragma unroll
    for (int r = 0; r < 8; ++r) {
      int k = a_kpair(r, half);
      b0.u[r] = (uint32_t)K0[(size_t)k*SEQ + s0] | ((uint32_t)K0[(size_t)(k+1)*SEQ + s0] << 16);
      b1.u[r] = (uint32_t)K1[(size_t)k*SEQ + s0] | ((uint32_t)K1[(size_t)(k+1)*SEQ + s0] << 16);
    }
    v8f c = {};
    c = __builtin_amdgcn_wmma_f32_16x16x32_bf16(false, q0.v, false, b0.v, (short)0, c, false, false);
    c = __builtin_amdgcn_wmma_f32_16x16x32_bf16(false, q1.v, false, b1.v, (short)0, c, false, false);
#pragma unroll
    for (int r = 0; r < 8; ++r)
      sc[(size_t)(r + 8*half)*SEQ + s0 + n15] = c[r] * SCALE;
  }
  __syncthreads();

  // ---- Phase 2: row softmax (8 threads per row, 256 cols each, float4 wide)
  int row = tid >> 3, seg = tid & 7;
  float* srow = sc + (size_t)row*SEQ + seg*256;
  float lmax = -3.4e38f;
  for (int i = 0; i < 256; i += 4) {
    float4 v = *(const float4*)(srow + i);
    lmax = fmaxf(lmax, fmaxf(fmaxf(v.x, v.y), fmaxf(v.z, v.w)));
  }
  red[tid] = lmax; __syncthreads();
  if (seg == 0) { float m = red[tid]; for (int j = 1; j < 8; ++j) m = fmaxf(m, red[tid+j]); red[tid] = m; }
  __syncthreads();
  float rmax = red[row*8];
  __syncthreads();
  float lsum = 0.f;
  for (int i = 0; i < 256; i += 4) {
    float4 v = *(const float4*)(srow + i);
    v.x = __expf(v.x - rmax); v.y = __expf(v.y - rmax);
    v.z = __expf(v.z - rmax); v.w = __expf(v.w - rmax);
    *(float4*)(srow + i) = v;
    lsum += (v.x + v.y) + (v.z + v.w);
  }
  red[tid] = lsum; __syncthreads();
  if (seg == 0) { float s = red[tid]; for (int j = 1; j < 8; ++j) s += red[tid+j]; red[tid] = s; }
  __syncthreads();
  float inv = 1.f / red[row*8];
  float*    wrow  = wout + ((size_t)bh*SEQ + t0 + row)*SEQ + seg*256;
  uint16_t* prow  = pb + (size_t)row*4096 + seg*512;   // packed bf16 slots
  for (int i = 0; i < 256; i += 4) {
    float4 v = *(const float4*)(srow + i);             // read BEFORE overwrite
    v.x *= inv; v.y *= inv; v.z *= inv; v.w *= inv;
    *(float4*)(wrow + i) = v;                          // b128 store of weights
    uint32_t p01 = (uint32_t)f2bf(v.x) | ((uint32_t)f2bf(v.y) << 16);
    uint32_t p23 = (uint32_t)f2bf(v.z) | ((uint32_t)f2bf(v.w) << 16);
    *(uint32_t*)(prow + i)     = p01;   // slots i..i+3 overwrite floats i/2, i/2+1 (< i) only
    *(uint32_t*)(prow + i + 2) = p23;
  }
  __syncthreads();

  // ---- Phase 3: O = P @ V ; wave owns 16 output columns d0..d0+15
  int d0 = wave*16;
  v8f o = {};
  const uint16_t* Prow = pb + (size_t)n15*4096;
  for (int kt = 0; kt < SEQ/32; ++kt) {
    int sbase = kt*32;
    int pidx  = ((sbase >> 8) << 9) + (sbase & 255);   // packed index of sbase
    const uint16_t* Vrow = Vb + ((size_t)bh*SEQ + sbase)*HD + d0 + n15;
    __builtin_prefetch(Vrow + 32*HD, 0, 0);            // next kt tile
    Frag16 a, bv;
#pragma unroll
    for (int r = 0; r < 8; ++r)
      a.u[r] = *(const uint32_t*)(Prow + pidx + a_kpair(r, half));
#pragma unroll
    for (int r = 0; r < 8; ++r) {
      int k = a_kpair(r, half);
      bv.u[r] = (uint32_t)Vrow[(size_t)k*HD] | ((uint32_t)Vrow[(size_t)(k+1)*HD] << 16);
    }
    o = __builtin_amdgcn_wmma_f32_16x16x32_bf16(false, a.v, false, bv.v, (short)0, o, false, false);
  }
#pragma unroll
  for (int r = 0; r < 8; ++r) {
    int t = t0 + r + 8*half;
    attnbf[((size_t)b*SEQ + t)*D_MODEL + h*HD + d0 + n15] = f2bf(o[r]);
  }
}

// ---------------------------------------------------------------- output projection
__global__ void __launch_bounds__(128)
out_gemm(const uint16_t* __restrict__ abf, const uint16_t* __restrict__ wbf,
         const float* __restrict__ bias, float* __restrict__ y) {
  int wave = threadIdx.x >> 5;
  int wtile = blockIdx.x*4 + wave;       // 256 m-tiles x 16 n-strips = 4096
  int tm = wtile & 255;
  int tn = wtile >> 8;
  int m0 = tm*16, n0 = tn*64;

  v8f acc[4] = {};
  gemm16x64_bf16(abf, wbf, D_MODEL, D_MODEL, m0, n0, D_MODEL, acc);

  int lane = threadIdx.x & 31;
  int n15 = lane & 15, half = lane >> 4;
#pragma unroll
  for (int j = 0; j < 4; ++j) {
    int n = n0 + j*16 + n15;
    float bj = bias[n];
#pragma unroll
    for (int r = 0; r < 8; ++r) {
      int m = m0 + r + 8*half;
      y[(size_t)m*D_MODEL + n] = acc[j][r] + bj;
    }
  }
}

// ---------------------------------------------------------------- launch
extern "C" void kernel_launch(void* const* d_in, const int* in_sizes, int n_in,
                              void* d_out, int out_size, void* d_ws, size_t ws_size,
                              hipStream_t stream) {
  (void)in_sizes; (void)n_in; (void)out_size; (void)ws_size;
  const float* x    = (const float*)d_in[0];
  const float* Wqkv = (const float*)d_in[1];
  const float* bqkv = (const float*)d_in[2];
  const float* Wo   = (const float*)d_in[3];
  const float* bo   = (const float*)d_in[4];

  float* y       = (float*)d_out;
  float* weights = (float*)d_out + (size_t)MROWS*D_MODEL;

  char* ws = (char*)d_ws;
  uint16_t* xbf    = (uint16_t*)(ws + 0);          //  8 MB  x bf16
  uint16_t* wqkvbf = (uint16_t*)(ws + 8388608);    //  6 MB  Wqkv bf16
  uint16_t* wobf   = (uint16_t*)(ws + 14680064);   //  2 MB  Wo bf16
  uint16_t* Qb     = (uint16_t*)(ws + 16777216);   //  8 MB  Q [bh,t,d]
  uint16_t* Ktr    = (uint16_t*)(ws + 25165824);   //  8 MB  K [bh,d,t]
  uint16_t* Vb     = (uint16_t*)(ws + 33554432);   //  8 MB  V [bh,t,d]
  uint16_t* attnbf = (uint16_t*)(ws + 41943040);   //  8 MB  attn out [b,t,D]

  cvt_bf16<<<2048, 256, 0, stream>>>(x,    xbf,    MROWS*D_MODEL);
  cvt_bf16<<<2048, 256, 0, stream>>>(Wqkv, wqkvbf, D_MODEL*NQKV);
  cvt_bf16<<<1024, 256, 0, stream>>>(Wo,   wobf,   D_MODEL*D_MODEL);

  qkv_gemm<<<3072, 128, 0, stream>>>(xbf, wqkvbf, bqkv, Qb, Ktr, Vb);

  size_t smem = (size_t)16*SEQ*sizeof(float);      // 128 KB -> 2 blocks/WGP
  attn_kernel<<<BATCH*NHEAD*(SEQ/16), 128, smem, stream>>>(Qb, Ktr, Vb, weights, attnbf);

  out_gemm<<<1024, 128, 0, stream>>>(attnbf, wobf, bo, y);
}
